// Morpher_78271484003233
// MI455X (gfx1250) — compile-verified, matrix operand order checked by reading, not run on hardware
//
#include <hip/hip_runtime.h>
#include <hip/hip_bf16.h>
#include <math.h>

typedef __bf16 bf16;
typedef __bf16 v16bf __attribute__((ext_vector_type(16)));
typedef float  v8f   __attribute__((ext_vector_type(8)));
typedef unsigned int u32;
typedef u32 u32x4 __attribute__((ext_vector_type(4)));

union Frag16 { v16bf v; u32x4 q[2]; };

// CDNA5 16-bit A/B operand fragment (16x32 tile, row-major source, stride in elems).
// lane<16: K = {0..7, 16..23}; lane>=16: K = {8..15, 24..31} (ISA 7.12.2).
__device__ __forceinline__ v16bf load_frag(const bf16* p, int ld, int kbase) {
  const int lane = threadIdx.x & 31;
  const int r = lane & 15;
  const int half = lane >> 4;
  const bf16* b = p + (size_t)r * ld + kbase + half * 8;
  Frag16 f;
  f.q[0] = *reinterpret_cast<const u32x4*>(b);
  f.q[1] = *reinterpret_cast<const u32x4*>(b + 16);
  return f.v;
}

__device__ __forceinline__ v8f wmma_bf16(v16bf a, v16bf b, v8f c) {
  return __builtin_amdgcn_wmma_f32_16x16x32_bf16(false, a, false, b, (short)0, c,
                                                 false, false);
}

__device__ __forceinline__ v8f vzero8() {
  v8f z = {0.f, 0.f, 0.f, 0.f, 0.f, 0.f, 0.f, 0.f};
  return z;
}

// CDNA5 async tensor path: per-lane global->LDS DMA, tracked by ASYNCcnt
// (ISA 08: GLOBAL_LOAD_ASYNC_TO_LDS_B128 / S_WAIT_ASYNCCNT). Low 32 bits of a
// generic LDS pointer are the LDS byte offset (flat aperture rule).
__device__ __forceinline__ void async_copy_b128(void* lds_dst, const void* gsrc) {
  unsigned lds = (unsigned)(size_t)lds_dst;
  asm volatile("global_load_async_to_lds_b128 %0, %1, off"
               :
               : "v"(lds), "v"(gsrc)
               : "memory");
}
__device__ __forceinline__ void async_wait0() {
  asm volatile("s_wait_asynccnt 0x0" ::: "memory");
}

// ---------------------------------------------------------------- converts ---
__global__ __launch_bounds__(256) void cvt_bf16(const float* __restrict__ src,
                                                bf16* __restrict__ dst, int n) {
  int i = blockIdx.x * 256 + threadIdx.x;
  if (i < n) dst[i] = (bf16)src[i];
}

// Wqkv (K,N,128,384) -> (K*N, 384, 128) : out-major rows for B-operand loads
__global__ __launch_bounds__(256) void cvt_wqkv(const float* __restrict__ src,
                                                bf16* __restrict__ dst) {
  int i = blockIdx.x * 256 + threadIdx.x;  // 12*128*384
  if (i >= 12 * 128 * 384) return;
  int e = i % 384;
  int rest = i / 384;
  int d = rest % 128;
  int kn = rest / 128;
  dst[((size_t)kn * 384 + e) * 128 + d] = (bf16)src[i];
}

// enc_B (64,4,896) -> (4, 896, 64)
__global__ __launch_bounds__(256) void cvt_encB(const float* __restrict__ src,
                                                bf16* __restrict__ dst) {
  int i = blockIdx.x * 256 + threadIdx.x;  // 64*4*896
  if (i >= 64 * 4 * 896) return;
  int d = i % 896;
  int rest = i / 896;
  int n = rest % 4;
  int r = rest / 4;
  dst[((size_t)n * 896 + d) * 64 + r] = (bf16)src[i];
}

// ---------------------------------------------------- generic WMMA NT GEMM ---
// C[M,N] = A[M,Kc] * W[N,Kc]^T ; bf16 in, f32 or bf16 out. Batched via grid.z.
template <bool OUT_BF16>
__global__ __launch_bounds__(256) void gemm_nt(
    const bf16* __restrict__ A, const bf16* __restrict__ W,
    void* __restrict__ Cout, int M, int N, int Kc,
    long sA, long sW, long sC) {
  __shared__ __attribute__((aligned(16))) bf16 As[64 * 40];
  __shared__ __attribute__((aligned(16))) bf16 Ws[64 * 40];
  const int batch = blockIdx.z;
  const bf16* Ab = A + (size_t)batch * sA;
  const bf16* Wb = W + (size_t)batch * sW;
  const int row0 = blockIdx.y * 64;
  const int col0 = blockIdx.x * 64;
  const int t = threadIdx.x;
  const int w = t >> 5;
  const int wm = w & 3, wn = w >> 2;
  const int lr = t >> 2;
  const int lc = (t & 3) * 8;
  v8f acc0 = vzero8(), acc1 = vzero8();
  for (int k0 = 0; k0 < Kc; k0 += 32) {
    __syncthreads();
    async_copy_b128(&As[lr * 40 + lc], &Ab[(size_t)(row0 + lr) * Kc + k0 + lc]);
    async_copy_b128(&Ws[lr * 40 + lc], &Wb[(size_t)(col0 + lr) * Kc + k0 + lc]);
    __builtin_prefetch(&Ab[(size_t)(row0 + lr) * Kc + k0 + 32 + lc], 0, 0);
    async_wait0();
    __syncthreads();
    v16bf a = load_frag(&As[(wm * 16) * 40], 40, 0);
    v16bf b0 = load_frag(&Ws[(wn * 32) * 40], 40, 0);
    v16bf b1 = load_frag(&Ws[(wn * 32 + 16) * 40], 40, 0);
    acc0 = wmma_bf16(a, b0, acc0);
    acc1 = wmma_bf16(a, b1, acc1);
  }
  const int lane = t & 31, r = lane & 15, half = lane >> 4;
#pragma unroll
  for (int v = 0; v < 8; ++v) {
    int m = row0 + wm * 16 + v + half * 8;
    int c = col0 + wn * 32 + r;
    if constexpr (OUT_BF16) {
      bf16* C = (bf16*)Cout + (size_t)batch * sC;
      C[(size_t)m * N + c] = (bf16)acc0[v];
      C[(size_t)m * N + c + 16] = (bf16)acc1[v];
    } else {
      float* C = (float*)Cout + (size_t)batch * sC;
      C[(size_t)m * N + c] = acc0[v];
      C[(size_t)m * N + c + 16] = acc1[v];
    }
  }
}

// -------------------------------------------------------------- encoder mix --
__global__ __launch_bounds__(256) void combine_z0(const float* __restrict__ base,
                                                  const float* __restrict__ delta,
                                                  const float* __restrict__ beta_p,
                                                  float* __restrict__ z0) {
  const float beta = *beta_p;
  long i = (long)blockIdx.x * 256 + threadIdx.x;  // 8192*4*896
  int d = (int)(i % 896);
  long rest = i / 896;
  int n = (int)(rest & 3);
  long m = rest >> 2;
  z0[i] = base[m * 896 + d] + beta * delta[((long)n * 8192 + m) * 896 + d];
}

// ------------------------------------------------ slot extract + head LN -----
__global__ __launch_bounds__(256) void slot_ln(const float* __restrict__ z0,
                                               const float* __restrict__ scale,
                                               const float* __restrict__ bias,
                                               const int* __restrict__ phase_p,
                                               bf16* __restrict__ xh) {
  int row = blockIdx.x * 256 + threadIdx.x;  // 12*8192 rows
  int z = row >> 13;
  int m = row & 8191;
  int n = z / 3, k = z % 3;
  int ph = *phase_p;
  const int soff_tab[3] = {0, 1, 3};
  const int sc[3] = {1, 2, 4};
  int act = soff_tab[k] + (ph % sc[k]);
  const float* src = z0 + ((size_t)m * 4 + n) * 896 + act * 128;
  float sum = 0.f, sq = 0.f;
#pragma unroll
  for (int i = 0; i < 32; ++i) {
    float4 vv = reinterpret_cast<const float4*>(src)[i];
    sum += vv.x + vv.y + vv.z + vv.w;
    sq += vv.x * vv.x + vv.y * vv.y + vv.z * vv.z + vv.w * vv.w;
  }
  float mu = sum * (1.f / 128.f);
  float var = sq * (1.f / 128.f) - mu * mu;
  float rs = rsqrtf(var + 1e-5f);
  bf16* dst = xh + (size_t)row * 128;
#pragma unroll
  for (int i = 0; i < 32; ++i) {
    float4 vv = reinterpret_cast<const float4*>(src)[i];
    int c = i * 4;
    dst[c + 0] = (bf16)((vv.x - mu) * rs * scale[c + 0] + bias[c + 0]);
    dst[c + 1] = (bf16)((vv.y - mu) * rs * scale[c + 1] + bias[c + 1]);
    dst[c + 2] = (bf16)((vv.z - mu) * rs * scale[c + 2] + bias[c + 2]);
    dst[c + 3] = (bf16)((vv.w - mu) * rs * scale[c + 3] + bias[c + 3]);
  }
}

// ----------------------------------------------------------------- QKV GEMM --
// grid (M/64, 12); block 128 (4 waves). Head (n,k) uses Wqkv[k, PERM[k,n]].
__global__ __launch_bounds__(128) void qkv_gemm(const bf16* __restrict__ Xh,
                                                const bf16* __restrict__ Wt,
                                                const int* __restrict__ phase_p,
                                                bf16* __restrict__ Q,
                                                bf16* __restrict__ Kb,
                                                bf16* __restrict__ V) {
  __shared__ __attribute__((aligned(16))) bf16 As[64 * 136];
  const int z = blockIdx.y;  // head = n*3+k
  const int n = z / 3, k = z % 3;
  const int ph = *phase_p;
  const int sc[3] = {1, 2, 4};
  int j = ph % sc[k];
  int alpha = 4 / sc[k];
  int psrc = (n + alpha * j) & 3;
  const bf16* Wkn = Wt + (size_t)(k * 4 + psrc) * 384 * 128;
  const int m0 = blockIdx.x * 64;
  const bf16* Asrc = Xh + ((size_t)z * 8192 + m0) * 128;
#pragma unroll
  for (int i = 0; i < 8; ++i) {
    int jj = threadIdx.x + i * 128;
    int row = jj >> 4, c = (jj & 15) * 8;
    async_copy_b128(&As[row * 136 + c], &Asrc[(size_t)row * 128 + c]);
  }
  async_wait0();
  __syncthreads();
  const int w = threadIdx.x >> 5;
  v16bf af[4];
#pragma unroll
  for (int ks = 0; ks < 4; ++ks) af[ks] = load_frag(&As[(w * 16) * 136], 136, ks * 32);
  const int lane = threadIdx.x & 31, r = lane & 15, half = lane >> 4;
  for (int et = 0; et < 24; ++et) {
    v8f acc = vzero8();
#pragma unroll
    for (int ks = 0; ks < 4; ++ks) {
      v16bf b = load_frag(Wkn + (size_t)(et * 16) * 128, 128, ks * 32);
      acc = wmma_bf16(af[ks], b, acc);
    }
    int colb = et * 16 + r;
    int sel = colb >> 7, c = colb & 127;
    bf16* dst = (sel == 0) ? Q : (sel == 1) ? Kb : V;
#pragma unroll
    for (int v = 0; v < 8; ++v) {
      int m = m0 + w * 16 + v + half * 8;
      int b_ = m >> 10, t_ = m & 1023;
      dst[(((size_t)b_ * 12 + z) * 1024 + t_) * 128 + c] = (bf16)acc[v];
    }
  }
}

// ------------------------------------------------------------ flash attn -----
// grid (T/64, B*12); block 128 (4 waves x 16 q-rows). RMW into z0 slice.
__global__ __launch_bounds__(128) void attn(const bf16* __restrict__ Q,
                                            const bf16* __restrict__ Kg,
                                            const bf16* __restrict__ Vg,
                                            const int* __restrict__ phase_p,
                                            float* __restrict__ z) {
  __shared__ __attribute__((aligned(16))) bf16 Ks[64 * 136];
  __shared__ __attribute__((aligned(16))) bf16 Vt[128 * 72];
  __shared__ __attribute__((aligned(16))) bf16 Pls[4 * 16 * 72];
  const int bh = blockIdx.y;
  const int b = bh / 12, h = bh % 12;
  const int n = h / 3, kk = h % 3;
  const int ph = *phase_p;
  const int soff_tab[3] = {0, 1, 3};
  const int sc[3] = {1, 2, 4};
  const int act = soff_tab[kk] + (ph % sc[kk]);
  const int qbase = blockIdx.x * 64;
  const int w = threadIdx.x >> 5;
  const int lane = threadIdx.x & 31, r = lane & 15, half = lane >> 4;
  const bf16* Qp = Q + ((size_t)bh * 1024 + qbase + w * 16) * 128;
  v16bf qf[4];
#pragma unroll
  for (int ks = 0; ks < 4; ++ks) qf[ks] = load_frag(Qp, 128, ks * 32);
  v8f O[8];
#pragma unroll
  for (int i = 0; i < 8; ++i) O[i] = vzero8();
  float mrun[8], lrun[8];
#pragma unroll
  for (int v = 0; v < 8; ++v) { mrun[v] = -1e30f; lrun[v] = 0.f; }
  const float scl = 0.08838834764831845f;  // 1/sqrt(128)
  for (int jb = 0; jb <= (int)blockIdx.x; ++jb) {
    __syncthreads();
    const bf16* Kp = Kg + ((size_t)bh * 1024 + jb * 64) * 128;
    const bf16* Vp = Vg + ((size_t)bh * 1024 + jb * 64) * 128;
#pragma unroll
    for (int i = 0; i < 8; ++i) {
      int jj = threadIdx.x + i * 128;
      int row = jj >> 4, c = (jj & 15) * 8;
      async_copy_b128(&Ks[row * 136 + c], &Kp[(size_t)row * 128 + c]);
      u32x4 vv = *reinterpret_cast<const u32x4*>(&Vp[(size_t)row * 128 + c]);
      const bf16* ve = reinterpret_cast<const bf16*>(&vv);
#pragma unroll
      for (int q = 0; q < 8; ++q) Vt[(c + q) * 72 + row] = ve[q];
    }
    async_wait0();
    __syncthreads();
    v8f S[4];
#pragma unroll
    for (int nt = 0; nt < 4; ++nt) {
      v8f s = vzero8();
#pragma unroll
      for (int ks = 0; ks < 4; ++ks) {
        v16bf bfr = load_frag(&Ks[(nt * 16) * 136], 136, ks * 32);
        s = wmma_bf16(qf[ks], bfr, s);
      }
      S[nt] = s;
    }
    const bool diag = (jb == (int)blockIdx.x);
#pragma unroll
    for (int nt = 0; nt < 4; ++nt)
#pragma unroll
      for (int v = 0; v < 8; ++v) {
        float val = S[nt][v] * scl;
        if (diag && (nt * 16 + r) > (w * 16 + v + half * 8)) val = -1e30f;
        S[nt][v] = val;
      }
#pragma unroll
    for (int v = 0; v < 8; ++v) {
      float mx = fmaxf(fmaxf(S[0][v], S[1][v]), fmaxf(S[2][v], S[3][v]));
#pragma unroll
      for (int msk = 1; msk < 16; msk <<= 1) mx = fmaxf(mx, __shfl_xor(mx, msk, 32));
      float mnew = fmaxf(mrun[v], mx);
      float corr = __expf(mrun[v] - mnew);
      mrun[v] = mnew;
      lrun[v] *= corr;
#pragma unroll
      for (int dt = 0; dt < 8; ++dt) O[dt][v] *= corr;
      float ls = 0.f;
#pragma unroll
      for (int nt = 0; nt < 4; ++nt) {
        float p = __expf(S[nt][v] - mnew);
        S[nt][v] = p;
        ls += p;
      }
#pragma unroll
      for (int msk = 1; msk < 16; msk <<= 1) ls += __shfl_xor(ls, msk, 32);
      lrun[v] += ls;
    }
    bf16* Pw = &Pls[w * 16 * 72];
#pragma unroll
    for (int nt = 0; nt < 4; ++nt)
#pragma unroll
      for (int v = 0; v < 8; ++v)
        Pw[(v + half * 8) * 72 + nt * 16 + r] = (bf16)S[nt][v];
    v16bf pa0 = load_frag(Pw, 72, 0);
    v16bf pa1 = load_frag(Pw, 72, 32);
#pragma unroll
    for (int dt = 0; dt < 8; ++dt) {
      v16bf b0 = load_frag(&Vt[(dt * 16) * 72], 72, 0);
      v16bf b1 = load_frag(&Vt[(dt * 16) * 72], 72, 32);
      O[dt] = wmma_bf16(pa0, b0, O[dt]);
      O[dt] = wmma_bf16(pa1, b1, O[dt]);
    }
  }
#pragma unroll
  for (int v = 0; v < 8; ++v) {
    float inv = 1.f / lrun[v];
    int m = b * 1024 + qbase + w * 16 + v + half * 8;
    float* dst = z + ((size_t)m * 4 + n) * 896 + act * 128;
#pragma unroll
    for (int dt = 0; dt < 8; ++dt) dst[dt * 16 + r] += O[dt][v] * inv;
  }
}

// --------------------------------------------------------- fused mixer MLP ---
// 32 rows/block, hidden in 64-wide chunks through LDS, persistent C accums.
__global__ __launch_bounds__(256) void mixer(const float* __restrict__ z1,
                                             const float* __restrict__ lns,
                                             const float* __restrict__ lnb,
                                             const bf16* __restrict__ W1,
                                             const float* __restrict__ b1v,
                                             const bf16* __restrict__ W2,
                                             const float* __restrict__ b2v,
                                             float* __restrict__ z2) {
  __shared__ __attribute__((aligned(16))) bf16 Xln[32 * 904];
  __shared__ __attribute__((aligned(16))) bf16 Hc[32 * 72];
  __shared__ float red1[256];
  __shared__ float red2[256];
  __shared__ float stats[64];
  const int rows0 = blockIdx.x * 32;
  const int t = threadIdx.x;
  const int row = t >> 3, seg = t & 7;
  const float* src = z1 + (size_t)(rows0 + row) * 896 + seg * 112;
  float sum = 0.f, sq = 0.f;
  for (int i = 0; i < 28; ++i) {
    float4 vv = reinterpret_cast<const float4*>(src)[i];
    sum += vv.x + vv.y + vv.z + vv.w;
    sq += vv.x * vv.x + vv.y * vv.y + vv.z * vv.z + vv.w * vv.w;
  }
  red1[t] = sum;
  red2[t] = sq;
  __syncthreads();
  if (t < 32) {
    float s = 0.f, q = 0.f;
    for (int i = 0; i < 8; ++i) { s += red1[t * 8 + i]; q += red2[t * 8 + i]; }
    float mu = s * (1.f / 896.f);
    float var = q * (1.f / 896.f) - mu * mu;
    stats[t * 2] = mu;
    stats[t * 2 + 1] = rsqrtf(var + 1e-5f);
  }
  __syncthreads();
  {
    float mu = stats[row * 2], rs = stats[row * 2 + 1];
    for (int i = 0; i < 28; ++i) {
      float4 vv = reinterpret_cast<const float4*>(src)[i];
      int c = seg * 112 + i * 4;
      Xln[row * 904 + c + 0] = (bf16)((vv.x - mu) * rs * lns[c + 0] + lnb[c + 0]);
      Xln[row * 904 + c + 1] = (bf16)((vv.y - mu) * rs * lns[c + 1] + lnb[c + 1]);
      Xln[row * 904 + c + 2] = (bf16)((vv.z - mu) * rs * lns[c + 2] + lnb[c + 2]);
      Xln[row * 904 + c + 3] = (bf16)((vv.w - mu) * rs * lns[c + 3] + lnb[c + 3]);
    }
  }
  __syncthreads();
  const int w = t >> 5;
  const int wm = w >> 2, wn = w & 3;   // big-C tiling: 2 x 4
  const int sm = w & 1, sn = w >> 1;   // hidden-chunk tiling: 2 x 4
  const int lane = t & 31, r = lane & 15, half = lane >> 4;
  v8f C[14];
#pragma unroll
  for (int i = 0; i < 14; ++i) C[i] = vzero8();
  for (int hc = 0; hc < 56; ++hc) {
    v8f s1 = vzero8();
    const bf16* W1p = W1 + (size_t)(hc * 64 + sn * 16) * 896;
    for (int ks = 0; ks < 28; ++ks) {
      v16bf a = load_frag(&Xln[(sm * 16) * 904], 904, ks * 32);
      v16bf bb = load_frag(W1p, 896, ks * 32);
      s1 = wmma_bf16(a, bb, s1);
    }
#pragma unroll
    for (int v = 0; v < 8; ++v) {
      int hcol = hc * 64 + sn * 16 + r;
      float x = s1[v] + b1v[hcol];
      float g = 0.5f * x * (1.f + erff(x * 0.7071067811865476f));
      Hc[(sm * 16 + v + half * 8) * 72 + sn * 16 + r] = (bf16)g;
    }
    __syncthreads();
    v16bf a0 = load_frag(&Hc[(wm * 16) * 72], 72, 0);
    v16bf a1 = load_frag(&Hc[(wm * 16) * 72], 72, 32);
#pragma unroll
    for (int dt = 0; dt < 14; ++dt) {
      const bf16* W2p = W2 + (size_t)(wn * 224 + dt * 16) * 3584 + hc * 64;
      v16bf b0 = load_frag(W2p, 3584, 0);
      v16bf b1f = load_frag(W2p, 3584, 32);
      C[dt] = wmma_bf16(a0, b0, C[dt]);
      C[dt] = wmma_bf16(a1, b1f, C[dt]);
    }
    __syncthreads();
  }
#pragma unroll
  for (int dt = 0; dt < 14; ++dt)
#pragma unroll
    for (int v = 0; v < 8; ++v) {
      int d = wn * 224 + dt * 16 + r;
      int m = rows0 + wm * 16 + v + half * 8;
      z2[(size_t)m * 896 + d] = z1[(size_t)m * 896 + d] + C[dt][v] + b2v[d];
    }
}

// ------------------------------------------------------------- decoder LN ----
__global__ __launch_bounds__(256) void dec_ln(const float* __restrict__ z2,
                                              const float* __restrict__ s,
                                              const float* __restrict__ bia,
                                              bf16* __restrict__ out) {
  __shared__ float red1[256];
  __shared__ float red2[256];
  const int m = blockIdx.x;
  const int t = threadIdx.x;
  const float* src = z2 + (size_t)m * 3584;
  float sum = 0.f, sq = 0.f;
  for (int i = t; i < 896; i += 256) {
    float4 vv = reinterpret_cast<const float4*>(src)[i];
    sum += vv.x + vv.y + vv.z + vv.w;
    sq += vv.x * vv.x + vv.y * vv.y + vv.z * vv.z + vv.w * vv.w;
  }
  red1[t] = sum;
  red2[t] = sq;
  __syncthreads();
  for (int st = 128; st > 0; st >>= 1) {
    if (t < st) { red1[t] += red1[t + st]; red2[t] += red2[t + st]; }
    __syncthreads();
  }
  float mu = red1[0] * (1.f / 3584.f);
  float var = red2[0] * (1.f / 3584.f) - mu * mu;
  float rs = rsqrtf(var + 1e-5f);
  bf16* dst = out + (size_t)m * 3584;
  for (int i = t; i < 896; i += 256) {
    float4 vv = reinterpret_cast<const float4*>(src)[i];
    int c = i * 4;
    dst[c + 0] = (bf16)((vv.x - mu) * rs * s[c + 0] + bia[c + 0]);
    dst[c + 1] = (bf16)((vv.y - mu) * rs * s[c + 1] + bia[c + 1]);
    dst[c + 2] = (bf16)((vv.z - mu) * rs * s[c + 2] + bia[c + 2]);
    dst[c + 3] = (bf16)((vv.w - mu) * rs * s[c + 3] + bia[c + 3]);
  }
}

// ------------------------------------------------------------------- host ----
extern "C" void kernel_launch(void* const* d_in, const int* in_sizes, int n_in,
                              void* d_out, int out_size, void* d_ws, size_t ws_size,
                              hipStream_t stream) {
  (void)in_sizes; (void)n_in; (void)out_size; (void)ws_size;
  const float* x          = (const float*)d_in[0];
  const float* enc_base_w = (const float*)d_in[1];
  const float* enc_A_w    = (const float*)d_in[2];
  const float* enc_B      = (const float*)d_in[3];
  const float* enc_beta   = (const float*)d_in[4];
  const float* WqkvF      = (const float*)d_in[5];
  const float* ln_attn_s  = (const float*)d_in[6];
  const float* ln_attn_b  = (const float*)d_in[7];
  const float* ln_mix_s   = (const float*)d_in[8];
  const float* ln_mix_b   = (const float*)d_in[9];
  const float* w1         = (const float*)d_in[10];
  const float* b1         = (const float*)d_in[11];
  const float* w2         = (const float*)d_in[12];
  const float* b2         = (const float*)d_in[13];
  const float* dls        = (const float*)d_in[14];
  const float* dlb        = (const float*)d_in[15];
  const float* ddw        = (const float*)d_in[16];
  const float* duw        = (const float*)d_in[17];
  const int*   phase      = (const int*)d_in[18];

  char* ws = (char*)d_ws;
  size_t off = 0;
  auto take = [&](size_t bytes) {
    size_t o = off;
    off = (off + bytes + 255) & ~(size_t)255;
    return o;
  };
  bf16*  xbf      = (bf16*)(ws + take(8192UL * 1024 * 2));
  bf16*  wbase_bf = (bf16*)(ws + take(896UL * 1024 * 2));
  bf16*  wA_bf    = (bf16*)(ws + take(64UL * 1024 * 2));
  bf16*  encBt_bf = (bf16*)(ws + take(4UL * 896 * 64 * 2));
  bf16*  wqkvt_bf = (bf16*)(ws + take(12UL * 384 * 128 * 2));
  bf16*  w1_bf    = (bf16*)(ws + take(3584UL * 896 * 2));
  bf16*  w2_bf    = (bf16*)(ws + take(896UL * 3584 * 2));
  bf16*  wdown_bf = (bf16*)(ws + take(64UL * 3584 * 2));
  bf16*  wup_bf   = (bf16*)(ws + take(1024UL * 64 * 2));
  float* basef    = (float*)(ws + take(8192UL * 896 * 4));
  bf16*  hbf      = (bf16*)(ws + take(8192UL * 64 * 2));
  float* deltaf   = (float*)(ws + take(4UL * 8192 * 896 * 4));
  float* z0f      = (float*)(ws + take(8192UL * 4 * 896 * 4));
  bf16*  xhead    = (bf16*)(ws + take(12UL * 8192 * 128 * 2));
  bf16*  Qb       = (bf16*)(ws + take(96UL * 1024 * 128 * 2));
  bf16*  Kb       = (bf16*)(ws + take(96UL * 1024 * 128 * 2));
  bf16*  Vb       = (bf16*)(ws + take(96UL * 1024 * 128 * 2));
  float* z2f      = (float*)(ws + take(32768UL * 896 * 4));
  bf16*  xdec     = (bf16*)(ws + take(8192UL * 3584 * 2));
  bf16*  hdec     = (bf16*)(ws + take(8192UL * 64 * 2));

  auto cvt = [&](const float* s, bf16* dd, long n) {
    cvt_bf16<<<dim3((unsigned)((n + 255) / 256)), dim3(256), 0, stream>>>(s, dd, (int)n);
  };
  cvt(x, xbf, 8192L * 1024);
  cvt(enc_base_w, wbase_bf, 896L * 1024);
  cvt(enc_A_w, wA_bf, 64L * 1024);
  cvt(w1, w1_bf, 3584L * 896);
  cvt(w2, w2_bf, 896L * 3584);
  cvt(ddw, wdown_bf, 64L * 3584);
  cvt(duw, wup_bf, 1024L * 64);
  cvt_wqkv<<<dim3(2304), dim3(256), 0, stream>>>(WqkvF, wqkvt_bf);
  cvt_encB<<<dim3(896), dim3(256), 0, stream>>>(enc_B, encBt_bf);

  // encoder: base / low-rank h / delta expansion
  gemm_nt<false><<<dim3(14, 128, 1), dim3(256), 0, stream>>>(
      xbf, wbase_bf, basef, 8192, 896, 1024, 0, 0, 0);
  gemm_nt<true><<<dim3(1, 128, 1), dim3(256), 0, stream>>>(
      xbf, wA_bf, hbf, 8192, 64, 1024, 0, 0, 0);
  gemm_nt<false><<<dim3(14, 128, 4), dim3(256), 0, stream>>>(
      hbf, encBt_bf, deltaf, 8192, 896, 64, 0, 896L * 64, 8192L * 896);
  combine_z0<<<dim3(114688), dim3(256), 0, stream>>>(basef, deltaf, enc_beta, z0f);

  // attention path
  slot_ln<<<dim3(384), dim3(256), 0, stream>>>(z0f, ln_attn_s, ln_attn_b, phase, xhead);
  qkv_gemm<<<dim3(128, 12), dim3(128), 0, stream>>>(xhead, wqkvt_bf, phase, Qb, Kb, Vb);
  attn<<<dim3(16, 96), dim3(128), 0, stream>>>(Qb, Kb, Vb, phase, z0f);

  // fused MLP mixer (z0f now holds z1)
  mixer<<<dim3(1024), dim3(256), 0, stream>>>(z0f, ln_mix_s, ln_mix_b, w1_bf, b1,
                                              w2_bf, b2, z2f);

  // decoder
  dec_ln<<<dim3(8192), dim3(256), 0, stream>>>(z2f, dls, dlb, xdec);
  gemm_nt<true><<<dim3(1, 128, 1), dim3(256), 0, stream>>>(
      xdec, wdown_bf, hdec, 8192, 64, 3584, 0, 0, 0);
  gemm_nt<false><<<dim3(16, 128, 1), dim3(256), 0, stream>>>(
      hdec, wup_bf, d_out, 8192, 1024, 64, 0, 0, 0);
}
